// WaveNet_41652592837472
// MI455X (gfx1250) — matrix operand order repaired and checked by reading, not compiled, not run on hardware
//
#include <hip/hip_runtime.h>
#include <hip/hip_bf16.h>

// ---------------------------------------------------------------------------
// WaveNet forward for MI455X (gfx1250, wave32, WMMA bf16 16x16x32, fp32 accum)
//
// Layouts:
//   h_f32, skip_sum : fp32, time-major [n][128],  n = b*T + t  (B=4, T=8192)
//   h_bf16          : bf16, time-major [n][128]   (matmul operand copy of h)
//   Prepped weights : bf16 A-operand tiles, [tile][lane(32)][16 bf16]
//       lane<16 : row M=lane,   K = kt*32 + {0..15}
//       lane>=16: row M=lane-16,K = kt*32 + {16..31}
//   WMMA B operand (from LDS, time-major): lane<16 -> N=lane, K=k0..k0+15
//       (16 consecutive channels = 32B contiguous); lane>=16 -> K=k0+16..k0+31
// ---------------------------------------------------------------------------

#define T_LEN 8192
#define BATCH 4
#define NPOS  (T_LEN * BATCH)   // 32768 positions
#define N_TILE 128
#define NUM_BLK 40

typedef __attribute__((ext_vector_type(16))) __bf16 v16bf;
typedef __attribute__((ext_vector_type(8)))  float  v8f;

union V16 { uint4 q[2]; v16bf v; };
union V8F { v8f v; float f[8]; };
union F4  { float4 q; float f[4]; };
union PK8 { __bf16 h[8]; uint4 q; };
union PK4 { __bf16 h[4]; uint2 q; };

#define WMMA_BF16(A, B, C) \
  __builtin_amdgcn_wmma_f32_16x16x32_bf16(false, (A), false, (B), (short)0, (C), false, false)

// ---------------------------------------------------------------------------
// One-time (per launch) weight conversion fp32 -> bf16 in WMMA A-operand order
// ---------------------------------------------------------------------------
__global__ __launch_bounds__(256) void prep_weights(
    const float* __restrict__ conv_w, const float* __restrict__ post_w,
    const float* __restrict__ lin1_w, const float* __restrict__ lin2_w,
    __bf16* __restrict__ Wc, __bf16* __restrict__ Wp,
    __bf16* __restrict__ Wl1, __bf16* __restrict__ Wl2) {
  const int CONV_JOBS = NUM_BLK * 2 * 16 * 4 * 32;  // 163840
  const int POST_JOBS = NUM_BLK * 8 * 4 * 32;       // 40960
  const int LIN_JOBS  = 2 * 8 * 4 * 32;             // 2048
  int j = blockIdx.x * 256 + threadIdx.x;
  if (j < CONV_JOBS) {
    int lane = j & 31, kt = (j >> 5) & 3, mt = (j >> 7) & 15;
    int tap = (j >> 11) & 1, blk = j >> 12;
    int co = mt * 16 + (lane & 15), khi = (lane >> 4) * 8;
    __bf16* dst = Wc + ((((size_t)blk * 2 + tap) * 16 + mt) * 4 + kt) * 512 + lane * 16;
#pragma unroll
    for (int e = 0; e < 16; ++e) {
      int ci = kt * 32 + e + khi * 2;
      dst[e] = (__bf16)conv_w[(((size_t)blk * 256 + co) * 128 + ci) * 2 + tap];
    }
  } else if (j < CONV_JOBS + POST_JOBS) {
    int j2 = j - CONV_JOBS;
    int lane = j2 & 31, kt = (j2 >> 5) & 3, mt = (j2 >> 7) & 7, blk = j2 >> 10;
    int co = mt * 16 + (lane & 15), khi = (lane >> 4) * 8;
    __bf16* dst = Wp + (((size_t)blk * 8 + mt) * 4 + kt) * 512 + lane * 16;
#pragma unroll
    for (int e = 0; e < 16; ++e) {
      int ci = kt * 32 + e + khi * 2;
      dst[e] = (__bf16)post_w[((size_t)blk * 128 + co) * 128 + ci];
    }
  } else if (j < CONV_JOBS + POST_JOBS + LIN_JOBS) {
    int j3 = j - CONV_JOBS - POST_JOBS;
    int lane = j3 & 31, kt = (j3 >> 5) & 3, mt = (j3 >> 7) & 7, which = (j3 >> 10) & 1;
    int co = mt * 16 + (lane & 15), khi = (lane >> 4) * 8;
    const float* src = which ? lin2_w : lin1_w;
    __bf16* dst = (which ? Wl2 : Wl1) + ((size_t)mt * 4 + kt) * 512 + lane * 16;
#pragma unroll
    for (int e = 0; e < 16; ++e) {
      int ci = kt * 32 + e + khi * 2;
      dst[e] = (__bf16)src[(size_t)co * 128 + ci];
    }
  }
}

// ---------------------------------------------------------------------------
// Pre-net: causal conv1d (1 -> 128, K=3), writes h (fp32+bf16), zeros skip_sum
// ---------------------------------------------------------------------------
__global__ __launch_bounds__(256) void prenet(
    const float* __restrict__ x, const float* __restrict__ pre_w,
    const float* __restrict__ pre_b,
    float* __restrict__ h_f32, __bf16* __restrict__ h_bf16,
    float* __restrict__ skip_sum) {
  int idx = blockIdx.x * 256 + threadIdx.x;  // NPOS * 32 threads
  int n = idx >> 5, c0 = (idx & 31) * 4;
  int b = n >> 13, t = n & (T_LEN - 1);
  const float* xb = x + (size_t)b * T_LEN;
  float xm2 = (t >= 2) ? xb[t - 2] : 0.f;
  float xm1 = (t >= 1) ? xb[t - 1] : 0.f;
  float x0 = xb[t];
  F4 hv; PK4 pk;
#pragma unroll
  for (int jj = 0; jj < 4; ++jj) {
    int c = c0 + jj;
    float v = pre_b[c] + pre_w[c * 3 + 0] * xm2 + pre_w[c * 3 + 1] * xm1 +
              pre_w[c * 3 + 2] * x0;
    hv.f[jj] = v; pk.h[jj] = (__bf16)v;
  }
  size_t off = (size_t)n * 128 + c0;
  *(float4*)(h_f32 + off) = hv.q;
  *(uint2*)(h_bf16 + off) = pk.q;
  float4 z; z.x = z.y = z.z = z.w = 0.f;
  *(float4*)(skip_sum + off) = z;
}

// ---------------------------------------------------------------------------
// One WaveNet residual block:
//   Y = W0 @ h[.,t-d] + W1 @ h[.,t] + cb ; skip = Y[0:128]*sigmoid(Y[128:256])
//   h += post_w @ skip + post_b ; skip_sum += skip
// Workgroup = 8 waves, tile = 128 time positions. Wave w owns out-channel pair
// (16w..16w+15 , 128+16w..128+16w+15).
// ---------------------------------------------------------------------------
__global__ __launch_bounds__(256) void wavenet_block(
    const __bf16* __restrict__ Wc_blk, const float* __restrict__ conv_b_blk,
    const __bf16* __restrict__ Wp_blk, const float* __restrict__ post_b_blk,
    float* __restrict__ h_f32, __bf16* __restrict__ h_bf16,
    float* __restrict__ skip_sum, int d) {
  __shared__ uint4 lds_u[128 * 17];  // unshifted h tile, 272B padded rows
  __shared__ uint4 lds_s[128 * 17];  // shifted h tile
  __shared__ uint4 lds_k[128 * 17];  // skip (bf16) tile
  const int tid = threadIdx.x;
  const int n0 = blockIdx.x * N_TILE;

  // Stage h (bf16) tiles: unshifted + causally shifted (zero when t < d)
#pragma unroll
  for (int it = 0; it < 8; ++it) {
    int ci = tid + it * 256;           // 2048 chunks of 16B
    int row = ci >> 4, colq = ci & 15; // 16 chunks per 256B row
    int n = n0 + row;
    lds_u[row * 17 + colq] = *((const uint4*)(h_bf16 + (size_t)n * 128) + colq);
    int t = n & (T_LEN - 1);
    uint4 z; z.x = z.y = z.z = z.w = 0u;
    if (t >= d)
      z = *((const uint4*)(h_bf16 + (size_t)(n - d) * 128) + colq);
    lds_s[row * 17 + colq] = z;
  }
  __syncthreads();

  const int w = tid >> 5, lane = tid & 31;
  const int lh = lane >> 4, lq = lane & 15;
  const int cbase = 16 * w + lh * 8;

  // Hoist conv A tiles (2 taps x {a,g} x 4 K-steps), 32B contiguous per lane
  v16bf A[2][2][4];
#pragma unroll
  for (int tap = 0; tap < 2; ++tap)
#pragma unroll
    for (int ag = 0; ag < 2; ++ag)
#pragma unroll
      for (int k = 0; k < 4; ++k)
        A[tap][ag][k] = *(const v16bf*)(
            Wc_blk + (((size_t)tap * 16 + (w + ag * 8)) * 4 + k) * 512 + lane * 16);

  F4 ba0, ba1, bg0, bg1;
  {
    const float* pa = conv_b_blk + cbase;
    ba0.q = *(const float4*)pa; ba1.q = *(const float4*)(pa + 4);
    const float* pg = conv_b_blk + 128 + cbase;
    bg0.q = *(const float4*)pg; bg1.q = *(const float4*)(pg + 4);
  }

  // GEMM1 + GLU + skip_sum, skip -> LDS
  for (int nt = 0; nt < 8; ++nt) {
    const int nrow = nt * 16 + lq;
    v8f acc_a = {0.f, 0.f, 0.f, 0.f, 0.f, 0.f, 0.f, 0.f};
    v8f acc_g = {0.f, 0.f, 0.f, 0.f, 0.f, 0.f, 0.f, 0.f};
#pragma unroll
    for (int k = 0; k < 4; ++k) {
      V16 Bu, Bs;
      int qi = nrow * 17 + k * 4 + lh * 2;
      Bu.q[0] = lds_u[qi]; Bu.q[1] = lds_u[qi + 1];
      Bs.q[0] = lds_s[qi]; Bs.q[1] = lds_s[qi + 1];
      acc_a = WMMA_BF16(A[0][0][k], Bs.v, acc_a);
      acc_a = WMMA_BF16(A[1][0][k], Bu.v, acc_a);
      acc_g = WMMA_BF16(A[0][1][k], Bs.v, acc_g);
      acc_g = WMMA_BF16(A[1][1][k], Bu.v, acc_g);
    }
    V8F av, gv; av.v = acc_a; gv.v = acc_g;
    float s[8];
#pragma unroll
    for (int r = 0; r < 8; ++r) {
      float a = av.f[r] + ((r < 4) ? ba0.f[r] : ba1.f[r - 4]);
      float g = gv.f[r] + ((r < 4) ? bg0.f[r] : bg1.f[r - 4]);
      s[r] = a * (1.0f / (1.0f + __expf(-g)));  // GLU
    }
    // skip_sum += skip (fp32 RMW, per-lane 32B contiguous, exclusive owner)
    size_t nsg = (size_t)(n0 + nrow);
    float* sp = skip_sum + nsg * 128 + cbase;
    F4 p0, p1; p0.q = *(float4*)sp; p1.q = *(float4*)(sp + 4);
#pragma unroll
    for (int r = 0; r < 4; ++r) { p0.f[r] += s[r]; p1.f[r] += s[r + 4]; }
    *(float4*)sp = p0.q; *(float4*)(sp + 4) = p1.q;
    // skip (bf16) -> LDS for GEMM2
    PK8 pk;
#pragma unroll
    for (int r = 0; r < 8; ++r) pk.h[r] = (__bf16)s[r];
    lds_k[nrow * 17 + (cbase >> 3)] = pk.q;
  }
  __syncthreads();

  // GEMM2: h += post_w @ skip + post_b ; refresh bf16 copy of h
  v16bf A2[4];
#pragma unroll
  for (int k = 0; k < 4; ++k)
    A2[k] = *(const v16bf*)(Wp_blk + ((size_t)w * 4 + k) * 512 + lane * 16);
  F4 pb0, pb1;
  {
    const float* pp = post_b_blk + cbase;
    pb0.q = *(const float4*)pp; pb1.q = *(const float4*)(pp + 4);
  }
  for (int nt = 0; nt < 8; ++nt) {
    const int nrow = nt * 16 + lq;
    v8f acc = {0.f, 0.f, 0.f, 0.f, 0.f, 0.f, 0.f, 0.f};
#pragma unroll
    for (int k = 0; k < 4; ++k) {
      V16 Bv; int qi = nrow * 17 + k * 4 + lh * 2;
      Bv.q[0] = lds_k[qi]; Bv.q[1] = lds_k[qi + 1];
      acc = WMMA_BF16(A2[k], Bv.v, acc);
    }
    V8F ac; ac.v = acc;
    size_t nsg = (size_t)(n0 + nrow);
    float* hp = h_f32 + nsg * 128 + cbase;
    F4 h0, h1; h0.q = *(float4*)hp; h1.q = *(float4*)(hp + 4);
    PK8 pk;
#pragma unroll
    for (int r = 0; r < 4; ++r) {
      float v0 = h0.f[r] + ac.f[r] + pb0.f[r];
      float v1 = h1.f[r] + ac.f[r + 4] + pb1.f[r];
      h0.f[r] = v0; h1.f[r] = v1;
      pk.h[r] = (__bf16)v0; pk.h[r + 4] = (__bf16)v1;
    }
    *(float4*)hp = h0.q; *(float4*)(hp + 4) = h1.q;
    *(uint4*)(h_bf16 + nsg * 128 + cbase) = pk.q;
  }
}

// ---------------------------------------------------------------------------
// Final: out[t,b,:] = relu(relu(skip_sum) @ lin1^T + b1) @ lin2^T + b2
// ---------------------------------------------------------------------------
__global__ __launch_bounds__(256) void wavenet_final(
    const float* __restrict__ skip_sum,
    const __bf16* __restrict__ Wl1, const float* __restrict__ lin1_b,
    const __bf16* __restrict__ Wl2, const float* __restrict__ lin2_b,
    float* __restrict__ out) {
  __shared__ uint4 lds_z0[128 * 17];
  __shared__ uint4 lds_z1[128 * 17];
  const int tid = threadIdx.x;
  const int n0 = blockIdx.x * N_TILE;

  // Stage relu(skip_sum) as bf16
#pragma unroll
  for (int it = 0; it < 8; ++it) {
    int ci = tid + it * 256;
    int row = ci >> 4, colq = ci & 15;
    const float* sp = skip_sum + ((size_t)(n0 + row)) * 128 + colq * 8;
    F4 f0, f1; f0.q = *(const float4*)sp; f1.q = *(const float4*)(sp + 4);
    PK8 pk;
#pragma unroll
    for (int r = 0; r < 4; ++r) {
      pk.h[r] = (__bf16)fmaxf(f0.f[r], 0.f);
      pk.h[r + 4] = (__bf16)fmaxf(f1.f[r], 0.f);
    }
    lds_z0[row * 17 + colq] = pk.q;
  }
  __syncthreads();

  const int w = tid >> 5, lane = tid & 31;
  const int lh = lane >> 4, lq = lane & 15;
  const int cbase = 16 * w + lh * 8;

  v16bf A1[4];
#pragma unroll
  for (int k = 0; k < 4; ++k)
    A1[k] = *(const v16bf*)(Wl1 + ((size_t)w * 4 + k) * 512 + lane * 16);
  F4 b10, b11;
  { const float* p = lin1_b + cbase; b10.q = *(const float4*)p; b11.q = *(const float4*)(p + 4); }

  for (int nt = 0; nt < 8; ++nt) {
    const int nrow = nt * 16 + lq;
    v8f acc = {0.f, 0.f, 0.f, 0.f, 0.f, 0.f, 0.f, 0.f};
#pragma unroll
    for (int k = 0; k < 4; ++k) {
      V16 Bv; int qi = nrow * 17 + k * 4 + lh * 2;
      Bv.q[0] = lds_z0[qi]; Bv.q[1] = lds_z0[qi + 1];
      acc = WMMA_BF16(A1[k], Bv.v, acc);
    }
    V8F ac; ac.v = acc;
    PK8 pk;
#pragma unroll
    for (int r = 0; r < 4; ++r) {
      pk.h[r] = (__bf16)fmaxf(ac.f[r] + b10.f[r], 0.f);
      pk.h[r + 4] = (__bf16)fmaxf(ac.f[r + 4] + b11.f[r], 0.f);
    }
    lds_z1[nrow * 17 + (cbase >> 3)] = pk.q;
  }
  __syncthreads();

  v16bf A2[4];
#pragma unroll
  for (int k = 0; k < 4; ++k)
    A2[k] = *(const v16bf*)(Wl2 + ((size_t)w * 4 + k) * 512 + lane * 16);
  F4 b20, b21;
  { const float* p = lin2_b + cbase; b20.q = *(const float4*)p; b21.q = *(const float4*)(p + 4); }

  for (int nt = 0; nt < 8; ++nt) {
    const int nrow = nt * 16 + lq;
    v8f acc = {0.f, 0.f, 0.f, 0.f, 0.f, 0.f, 0.f, 0.f};
#pragma unroll
    for (int k = 0; k < 4; ++k) {
      V16 Bv; int qi = nrow * 17 + k * 4 + lh * 2;
      Bv.q[0] = lds_z1[qi]; Bv.q[1] = lds_z1[qi + 1];
      acc = WMMA_BF16(A2[k], Bv.v, acc);
    }
    V8F ac; ac.v = acc;
    int n = n0 + nrow;
    int b = n >> 13, t = n & (T_LEN - 1);
    float* op = out + ((size_t)t * BATCH + b) * 128 + cbase;  // [T,B,C]
    F4 o0, o1;
#pragma unroll
    for (int r = 0; r < 4; ++r) {
      o0.f[r] = ac.f[r] + b20.f[r];
      o1.f[r] = ac.f[r + 4] + b21.f[r];
    }
    *(float4*)op = o0.q; *(float4*)(op + 4) = o1.q;
  }
}

// ---------------------------------------------------------------------------
extern "C" void kernel_launch(void* const* d_in, const int* in_sizes, int n_in,
                              void* d_out, int out_size, void* d_ws, size_t ws_size,
                              hipStream_t stream) {
  const float* x      = (const float*)d_in[0];
  const float* pre_w  = (const float*)d_in[1];
  const float* pre_b  = (const float*)d_in[2];
  const float* conv_w = (const float*)d_in[3];
  const float* conv_b = (const float*)d_in[4];
  const float* post_w = (const float*)d_in[5];
  const float* post_b = (const float*)d_in[6];
  const float* lin1_b = (const float*)d_in[8];
  const float* lin1_w = (const float*)d_in[7];
  const float* lin2_w = (const float*)d_in[9];
  const float* lin2_b = (const float*)d_in[10];
  float* out = (float*)d_out;

  char* ws = (char*)d_ws;
  size_t off = 0;
  auto alloc = [&](size_t bytes) {
    void* p = ws + off;
    off = (off + bytes + 255) & ~(size_t)255;
    return p;
  };
  float*  h_f32    = (float*) alloc((size_t)NPOS * 128 * 4);   // 16 MB
  __bf16* h_bf16   = (__bf16*)alloc((size_t)NPOS * 128 * 2);   // 8 MB
  float*  skip_sum = (float*) alloc((size_t)NPOS * 128 * 4);   // 16 MB
  __bf16* Wc  = (__bf16*)alloc((size_t)NUM_BLK * 2 * 256 * 128 * 2);  // 5.25 MB
  __bf16* Wp  = (__bf16*)alloc((size_t)NUM_BLK * 128 * 128 * 2);      // 1.3 MB
  __bf16* Wl1 = (__bf16*)alloc((size_t)128 * 128 * 2);
  __bf16* Wl2 = (__bf16*)alloc((size_t)128 * 128 * 2);

  prep_weights<<<808, 256, 0, stream>>>(conv_w, post_w, lin1_w, lin2_w,
                                        Wc, Wp, Wl1, Wl2);
  prenet<<<(NPOS * 32) / 256, 256, 0, stream>>>(x, pre_w, pre_b,
                                                h_f32, h_bf16, skip_sum);
  for (int i = 0; i < NUM_BLK; ++i) {
    int d = 1 << (i % 10);
    wavenet_block<<<NPOS / N_TILE, 256, 0, stream>>>(
        Wc + (size_t)i * 2 * 16 * 4 * 512, conv_b + i * 256,
        Wp + (size_t)i * 8 * 4 * 512, post_b + i * 128,
        h_f32, h_bf16, skip_sum, d);
  }
  wavenet_final<<<NPOS / N_TILE, 256, 0, stream>>>(skip_sum, Wl1, lin1_b,
                                                   Wl2, lin2_b, out);
}
